// GNUMGraph_33749853012156
// MI455X (gfx1250) — matrix operand order repaired and checked by reading, not compiled, not run on hardware
//
#include <hip/hip_runtime.h>
#include <hip/hip_bf16.h>
#include <math.h>

// ---------------------------------------------------------------------------
// Types for CDNA5 WMMA (wave32): V_WMMA_F32_16X16X32_BF16
// ---------------------------------------------------------------------------
typedef __bf16 bf16_t;
typedef __attribute__((ext_vector_type(16))) __bf16 v16bf;
typedef __attribute__((ext_vector_type(8)))  float  v8f;

#define IN_DIM 128   // all GEMMs here are [M,128] @ [128,128]

__device__ inline float bf2f(unsigned int u16bits) {
  union { unsigned int i; float f; } x;
  x.i = u16bits << 16;
  return x.f;
}

// ---------------------------------------------------------------------------
// GEMM: C[M,128] = A[M,128] @ W[128,128]  (bf16 WMMA, f32 accumulate,
// bf16 output for the L2-resident gather stream).
// Block = 256 threads = 8 waves, covers 64 rows x 128 cols.
// Wave w owns column tile [16w,16w+16) across 4 row tiles; K=128 in 4 steps.
// ---------------------------------------------------------------------------
__global__ void __launch_bounds__(256)
gemm128_bf16_wmma(const float* __restrict__ A, const float* __restrict__ W,
                  bf16_t* __restrict__ C, int M) {
  __shared__ __align__(32) bf16_t ldsA[64 * 128];    // A tile / C staging (16 KB)
  __shared__ __align__(32) bf16_t ldsWt[128 * 128];  // W transposed [n][k] (32 KB)

  const int tid  = threadIdx.x;
  const int row0 = blockIdx.x * 64;

  // Stage A tile (f32 -> bf16), row-major [r][k]
  for (int i = tid; i < 64 * 128; i += 256) {
    int r = i >> 7, k = i & 127;
    float v = (row0 + r < M) ? A[(size_t)(row0 + r) * 128 + k] : 0.0f;
    ldsA[i] = (bf16_t)v;
  }
  // Stage W transposed (f32 -> bf16): W is [k][n] row-major in memory
  for (int i = tid; i < 128 * 128; i += 256) {
    int k = i >> 7, n = i & 127;
    ldsWt[n * 128 + k] = (bf16_t)W[i];
  }
  __syncthreads();

  const int wv   = tid >> 5;
  const int lane = tid & 31;
  const int half = lane >> 4;   // lane-half per ISA 16-bit operand layout
  const int m    = lane & 15;
  const int n    = wv * 16 + m;

  v8f c[4];
  #pragma unroll
  for (int rt = 0; rt < 4; ++rt) c[rt] = (v8f){0.f,0.f,0.f,0.f,0.f,0.f,0.f,0.f};

  #pragma unroll
  for (int kk = 0; kk < 128; kk += 32) {
    // B fragment: 32x16 bf16; lane-half selects K block of 16, contiguous in
    // transposed LDS -> 32B vector load. Reused across 4 row tiles.
    v16bf b = *(const v16bf*)&ldsWt[n * 128 + kk + half * 16];
    #pragma unroll
    for (int rt = 0; rt < 4; ++rt) {
      // A fragment: 16x32 bf16, contiguous K order per lane half
      // (element e -> K = half*16 + e, matching B's contiguous K order)
      v16bf a;
      #pragma unroll
      for (int e = 0; e < 16; ++e) {
        int kA = half * 16 + e;
        a[e] = ldsA[(rt * 16 + m) * 128 + kk + kA];
      }
      c[rt] = __builtin_amdgcn_wmma_f32_16x16x32_bf16(
          /*neg_a=*/false, a, /*neg_b=*/false, b,
          /*c_mod=*/(short)0, c[rt], /*reuse_a=*/false, /*reuse_b=*/false);
    }
  }

  // Stage C (bf16) back through LDS so global stores are coalesced b128.
  __syncthreads();   // everyone done reading ldsA
  #pragma unroll
  for (int rt = 0; rt < 4; ++rt) {
    // C/D layout: VGPR r -> M = r + half*8 (within 16-row tile), N = lane&15
    #pragma unroll
    for (int r = 0; r < 8; ++r)
      ldsA[(rt * 16 + half * 8 + r) * 128 + n] = (bf16_t)c[rt][r];
  }
  __syncthreads();

  // 64*128 bf16 = 16 KB: 1024 chunks of 16B, 4 per thread, fully coalesced.
  uint4* dst = (uint4*)(C + (size_t)row0 * 128);
  const uint4* src = (const uint4*)ldsA;
  for (int i = tid; i < 1024; i += 256) {
    int row = row0 + (i >> 4);
    if (row < M) dst[i] = src[i];
  }
}

// ---------------------------------------------------------------------------
// Zero a float buffer (graph-capture-safe)
// ---------------------------------------------------------------------------
__global__ void zero_f32(float* __restrict__ p, long long n4) {
  long long i = (long long)blockIdx.x * blockDim.x + threadIdx.x;
  if (i < n4) ((float4*)p)[i] = make_float4(0.f, 0.f, 0.f, 0.f);
}

// ---------------------------------------------------------------------------
// Edge scatter: agg[dst] += w * support[src]; one wave per edge.
// support is bf16 (L2-resident, half the gather bytes); agg stays f32 with
// hardware global_atomic_add_f32.
// ---------------------------------------------------------------------------
__global__ void __launch_bounds__(256)
edge_scatter(const bf16_t* __restrict__ support, const float* __restrict__ ew,
             const int* __restrict__ esrc, const int* __restrict__ edst,
             float* __restrict__ agg, int E) {
  int e = blockIdx.x * 8 + (threadIdx.x >> 5);
  if (e >= E) return;
  int lane = threadIdx.x & 31;
  int s = esrc[e], d = edst[e];
  float w = ew[e];
  // 4 bf16 features per lane (8 bytes)
  uint2 p = ((const uint2*)(support + (size_t)s * 128))[lane];
  float v0 = bf2f(p.x & 0xffffu), v1 = bf2f(p.x >> 16);
  float v2 = bf2f(p.y & 0xffffu), v3 = bf2f(p.y >> 16);
  float* out = agg + (size_t)d * 128 + lane * 4;
  unsafeAtomicAdd(out + 0, v0 * w);
  unsafeAtomicAdd(out + 1, v1 * w);
  unsafeAtomicAdd(out + 2, v2 * w);
  unsafeAtomicAdd(out + 3, v3 * w);
}

// ---------------------------------------------------------------------------
// h = relu(h + b)  (in place, b broadcast over feature dim 128)
// ---------------------------------------------------------------------------
__global__ void bias_relu(float* __restrict__ h, const float* __restrict__ b,
                          long long total) {
  long long i = (long long)blockIdx.x * blockDim.x + threadIdx.x;
  if (i < total) {
    float v = h[i] + b[i & 127];
    h[i] = v > 0.f ? v : 0.f;
  }
}

// ---------------------------------------------------------------------------
// Fused heads: tau = relu(rep@tw1 + tb1)@tw2 + tb2 ; e = sigmoid(rep@pw + pb)
// One wave per node. Each lane owns hidden units j = lane and j = lane+32:
// tw1 reads are lane-consecutive (bank-conflict-free); rep[d] broadcast via
// shuffles (128 total); a single wave reduction at the end.
// ---------------------------------------------------------------------------
__device__ inline float wave_sum(float v) {
  #pragma unroll
  for (int m = 16; m >= 1; m >>= 1) v += __shfl_xor(v, m, 32);
  return v;
}

__global__ void __launch_bounds__(256)
heads_kernel(const float* __restrict__ rep,
             const float* __restrict__ tw1, const float* __restrict__ tb1,
             const float* __restrict__ tw2, const float* __restrict__ tb2,
             const float* __restrict__ pw,  const float* __restrict__ pb,
             float* __restrict__ tau_out, float* __restrict__ e_out, int N) {
  __shared__ float s_tw1[128 * 64];   // [d][j], 32 KB
  __shared__ float s_tb1[64];
  __shared__ float s_tw2[64];
  __shared__ float s_pw[128];

  const int tid = threadIdx.x;
  for (int i = tid; i < 128 * 64; i += 256) s_tw1[i] = tw1[i];
  if (tid < 64)  { s_tb1[tid] = tb1[tid]; s_tw2[tid] = tw2[tid]; }
  if (tid < 128) s_pw[tid] = pw[tid];
  __syncthreads();

  int node = blockIdx.x * 8 + (tid >> 5);
  if (node >= N) return;
  int lane = tid & 31;

  float4 r4 = ((const float4*)(rep + (size_t)node * 128))[lane];
  const int d0 = lane * 4;

  float acc0 = 0.f, acc1 = 0.f;   // hidden units j = lane, j = lane+32
  #pragma unroll 8
  for (int dd = 0; dd < 32; ++dd) {
    float rx = __shfl(r4.x, dd, 32);
    float ry = __shfl(r4.y, dd, 32);
    float rz = __shfl(r4.z, dd, 32);
    float rw = __shfl(r4.w, dd, 32);
    const float* w0 = &s_tw1[(dd * 4) * 64 + lane];
    acc0 += rx * w0[0]   + ry * w0[64]   + rz * w0[128]   + rw * w0[192];
    acc1 += rx * w0[32]  + ry * w0[96]   + rz * w0[160]   + rw * w0[224];
  }
  float h0 = acc0 + s_tb1[lane];        h0 = h0 > 0.f ? h0 : 0.f;
  float h1 = acc1 + s_tb1[lane + 32];   h1 = h1 > 0.f ? h1 : 0.f;
  float tau_part = h0 * s_tw2[lane] + h1 * s_tw2[lane + 32];

  float q_part = r4.x * s_pw[d0 + 0] + r4.y * s_pw[d0 + 1]
               + r4.z * s_pw[d0 + 2] + r4.w * s_pw[d0 + 3];

  float tau = wave_sum(tau_part) + tb2[0];
  float q   = wave_sum(q_part)   + pb[0];
  float sig = 1.0f / (1.0f + expf(-q));

  if (lane == 0) {
    tau_out[node] = tau;
    e_out[node]   = sig;
  }
}

// ---------------------------------------------------------------------------
// Launcher
// ---------------------------------------------------------------------------
extern "C" void kernel_launch(void* const* d_in, const int* in_sizes, int n_in,
                              void* d_out, int out_size, void* d_ws, size_t ws_size,
                              hipStream_t stream) {
  const float* x   = (const float*)d_in[0];
  const float* ew  = (const float*)d_in[1];
  const float* W1  = (const float*)d_in[2];
  const float* b1  = (const float*)d_in[3];
  const float* W2  = (const float*)d_in[4];
  const float* b2  = (const float*)d_in[5];
  const float* tw1 = (const float*)d_in[6];
  const float* tb1 = (const float*)d_in[7];
  const float* tw2 = (const float*)d_in[8];
  const float* tb2 = (const float*)d_in[9];
  const float* pw  = (const float*)d_in[10];
  const float* pb  = (const float*)d_in[11];
  const int* esrc  = (const int*)d_in[12];
  const int* edst  = (const int*)d_in[13];

  const int N = in_sizes[0] / IN_DIM;
  const int E = in_sizes[1];
  const long long NF = (long long)N * IN_DIM;

  float* tau = (float*)d_out;
  float* eo  = tau + N;
  float* rep = tau + 2 * (size_t)N;

  float*  agg1 = (float*)d_ws;                       // f32 agg/h1 [N,128]
  bf16_t* sup  = (bf16_t*)(agg1 + (size_t)N * 128);  // bf16 support [N,128]

  const int gemm_blocks = (N + 63) / 64;
  const int edge_blocks = (E + 7) / 8;
  const int ew_blocks   = (int)((NF + 255) / 256);
  const int zero_blocks = (int)((NF / 4 + 255) / 256);
  const int head_blocks = (N + 7) / 8;

  // ---- Layer 1 ----
  gemm128_bf16_wmma<<<gemm_blocks, 256, 0, stream>>>(x, W1, sup, N);
  zero_f32<<<zero_blocks, 256, 0, stream>>>(agg1, NF / 4);
  edge_scatter<<<edge_blocks, 256, 0, stream>>>(sup, ew, esrc, edst, agg1, E);
  bias_relu<<<ew_blocks, 256, 0, stream>>>(agg1, b1, NF);

  // ---- Layer 2 (aggregate directly into rep region of d_out) ----
  gemm128_bf16_wmma<<<gemm_blocks, 256, 0, stream>>>(agg1, W2, sup, N);
  zero_f32<<<zero_blocks, 256, 0, stream>>>(rep, NF / 4);
  edge_scatter<<<edge_blocks, 256, 0, stream>>>(sup, ew, esrc, edst, rep, E);
  bias_relu<<<ew_blocks, 256, 0, stream>>>(rep, b2, NF);

  // ---- Heads ----
  heads_kernel<<<head_blocks, 256, 0, stream>>>(rep, tw1, tb1, tw2, tb2,
                                                pw, pb, tau, eo, N);
}
